// SensorAttention_46076409152184
// MI455X (gfx1250) — compile-verified, matrix operand order checked by reading, not run on hardware
//
#include <hip/hip_runtime.h>
#include <hip/hip_bf16.h>
#include <math.h>

#define T_DIM 1024
#define B_DIM 512
#define D_DIM 128

typedef __attribute__((ext_vector_type(16))) __bf16 v16bf;
typedef __attribute__((ext_vector_type(8)))  __bf16 v8bf;
typedef __attribute__((ext_vector_type(8)))  float  v8f;

// A/B 16-bit fragment load (16x32 A or 32x16 B tile slice for one lane).
// Per ISA layout: lane half selects K+8 offset; elements 0..7 -> K=base..base+7,
// elements 8..15 -> K=base+16..base+23. Two contiguous 16B LDS reads.
__device__ __forceinline__ v16bf load_frag(const __bf16* rowbase, int kbase) {
  v16bf r;
  v8bf lo = *(const v8bf*)(rowbase + kbase);
  v8bf hi = *(const v8bf*)(rowbase + kbase + 16);
#pragma unroll
  for (int e = 0; e < 8; ++e) { r[e] = lo[e]; r[e + 8] = hi[e]; }
  return r;
}

__device__ __forceinline__ float fast_tanh(float v) {
#if __has_builtin(__builtin_amdgcn_tanhf)
  return __builtin_amdgcn_tanhf(v);   // V_TANH_F32 (CDNA5 trans op)
#else
  return tanhf(v);
#endif
}

__device__ __forceinline__ float fast_rcp(float v) {
#if __has_builtin(__builtin_amdgcn_rcpf)
  return __builtin_amdgcn_rcpf(v);
#else
  return 1.0f / v;
#endif
}

// Async global->LDS copy of one 16-byte chunk (GLOBAL_LOAD_ASYNC_TO_LDS_B128,
// tracked by ASYNCcnt). LDS operand is the wave-relative LDS byte address
// (low 32 bits of the flat shared address).
__device__ __forceinline__ void async_copy_b128(void* lds_dst, const float* gsrc) {
  unsigned lds_off = (unsigned)(uintptr_t)lds_dst;
  asm volatile("global_load_async_to_lds_b128 %0, %1, off"
               :: "v"(lds_off), "v"(gsrc)
               : "memory");
}

__device__ __forceinline__ void wait_async0() {
  asm volatile("s_wait_asynccnt 0x0" ::: "memory");
}

// Stage a full 16x128 fp32 tile (8KB = 512 x 16B chunks) with 2 chunks/thread.
__device__ __forceinline__ void stage_x_async(const float* xt, float* dst, int tid) {
  async_copy_b128(dst + tid * 4,         xt + tid * 4);
  async_copy_b128(dst + (tid + 256) * 4, xt + (tid + 256) * 4);
}

__global__ __launch_bounds__(256) void sensor_attention_scan(
    const float* __restrict__ x,  const float* __restrict__ Wx,
    const float* __restrict__ bx, const float* __restrict__ Wb,
    const float* __restrict__ bb, const float* __restrict__ We,
    const float* __restrict__ be,
    float* __restrict__ out_sig,   // [T,B,D]
    float* __restrict__ out_beta)  // [B,T,D]
{
  __shared__ float  s_x2[2][16 * 128]; // double-buffered x_t fp32 (async dest)
  __shared__ __bf16 s_xb[16 * 128];    // x_t bf16 (A operand)
  __shared__ __bf16 s_beta[16 * 128];  // beta_{t-1} bf16 (A operand)
  __shared__ __bf16 s_mrg[16 * 128];   // tanh(...) bf16 (A operand)
  __shared__ float  s_en[16 * 128];    // energy / exp scratch fp32
  __shared__ float  s_rmx[16 * 16];    // softmax max partials
  __shared__ float  s_rsm[16 * 16];    // softmax sum partials

  const int tid   = threadIdx.x;
  const int lane  = tid & 31;
  const int wv    = tid >> 5;          // wave 0..7 -> N tile
  const int m0    = blockIdx.x * 16;   // batch-row tile base
  const int n0    = wv * 16;
  const int bn    = lane & 15;         // N within tile (B frag / C-D col)
  const int khalf = (lane >> 4) << 3;  // lane-half K offset (0 or 8)
  const int arow  = lane & 15;         // A-frag row M
  const int crow  = (lane >> 4) << 3;  // C/D row offset (0 or 8)

  // Kick off async staging of x_0 before touching weights (overlaps HBM latency).
  stage_x_async(x + (size_t)m0 * D_DIM, s_x2[0], tid);

  // ---- prologue: weights -> register-resident bf16 B-fragments (loop-invariant) ----
  // B[k][n] = W[n][k] (computing x @ W^T), read row-major from global fp32.
  v16bf fWx[4], fWb[4], fWe[4];
#pragma unroll
  for (int kb = 0; kb < 4; ++kb) {
    const int kbase = kb * 32 + khalf;
    const float* rx = Wx + (n0 + bn) * D_DIM + kbase;
    const float* rb = Wb + (n0 + bn) * D_DIM + kbase;
    const float* re = We + (n0 + bn) * D_DIM + kbase;
#pragma unroll
    for (int e = 0; e < 8; ++e) {
      fWx[kb][e] = (__bf16)rx[e];  fWx[kb][e + 8] = (__bf16)rx[e + 16];
      fWb[kb][e] = (__bf16)rb[e];  fWb[kb][e + 8] = (__bf16)rb[e + 16];
      fWe[kb][e] = (__bf16)re[e];  fWe[kb][e + 8] = (__bf16)re[e + 16];
    }
  }
  const float bias1 = bx[n0 + bn] + bb[n0 + bn];  // fused bias for GEMM1+2
  const float bias2 = be[n0 + bn];

  for (int i = tid; i < 16 * 128; i += 256) s_beta[i] = (__bf16)0.0f; // beta0 = 0

  const int r  = tid >> 4;   // softmax: row 0..15
  const int cg = tid & 15;   // softmax: column group

  for (int t = 0; t < T_DIM; ++t) {
    float* xf = s_x2[t & 1];

    // ---- x_t tile ready in LDS; also quiesces all prior-step LDS traffic ----
    wait_async0();
    __syncthreads();

    // Prefetch x_{t+1} into the other buffer (ASYNCcnt-tracked, overlaps compute).
    if (t + 1 < T_DIM)
      stage_x_async(x + (size_t)(t + 1) * (B_DIM * D_DIM) + (size_t)m0 * D_DIM,
                    s_x2[(t + 1) & 1], tid);

    // fp32 -> bf16 conversion pass (8 contiguous elements / thread, one 16B store)
    {
      const float4 a = *(const float4*)(xf + tid * 8);
      const float4 b = *(const float4*)(xf + tid * 8 + 4);
      v8bf o;
      o[0] = (__bf16)a.x; o[1] = (__bf16)a.y; o[2] = (__bf16)a.z; o[3] = (__bf16)a.w;
      o[4] = (__bf16)b.x; o[5] = (__bf16)b.y; o[6] = (__bf16)b.z; o[7] = (__bf16)b.w;
      *(v8bf*)(&s_xb[tid * 8]) = o;
    }
    __syncthreads();

    // ---- merged = tanh(x@Wx^T + beta@Wb^T + bx + bb)  (fused dual GEMM) ----
    v8f acc;
#pragma unroll
    for (int i = 0; i < 8; ++i) acc[i] = bias1;
#pragma unroll
    for (int kb = 0; kb < 4; ++kb) {
      v16bf a1 = load_frag(&s_xb[arow * 128], kb * 32 + khalf);
      acc = __builtin_amdgcn_wmma_f32_16x16x32_bf16(false, a1, false, fWx[kb],
                                                    (short)0, acc, false, false);
      v16bf a2 = load_frag(&s_beta[arow * 128], kb * 32 + khalf);
      acc = __builtin_amdgcn_wmma_f32_16x16x32_bf16(false, a2, false, fWb[kb],
                                                    (short)0, acc, false, false);
    }
#pragma unroll
    for (int i = 0; i < 8; ++i)
      s_mrg[(crow + i) * 128 + n0 + bn] = (__bf16)fast_tanh(acc[i]);
    __syncthreads();

    // ---- energy = merged @ We^T + be ----
    v8f acc2;
#pragma unroll
    for (int i = 0; i < 8; ++i) acc2[i] = bias2;
#pragma unroll
    for (int kb = 0; kb < 4; ++kb) {
      v16bf a3 = load_frag(&s_mrg[arow * 128], kb * 32 + khalf);
      acc2 = __builtin_amdgcn_wmma_f32_16x16x32_bf16(false, a3, false, fWe[kb],
                                                     (short)0, acc2, false, false);
    }
#pragma unroll
    for (int i = 0; i < 8; ++i)
      s_en[(crow + i) * 128 + n0 + bn] = acc2[i];
    __syncthreads();

    // ---- softmax over D=128 per row (fp32), two reduction buffers, 2 barriers ----
    float mx = -1e30f;
#pragma unroll
    for (int k = 0; k < 8; ++k) mx = fmaxf(mx, s_en[r * 128 + cg + 16 * k]);
    s_rmx[r * 16 + cg] = mx;
    __syncthreads();

    float rmax = -1e30f;
#pragma unroll
    for (int j = 0; j < 16; ++j) rmax = fmaxf(rmax, s_rmx[r * 16 + j]);
    float ps = 0.0f;
#pragma unroll
    for (int k = 0; k < 8; ++k) {
      const int c = cg + 16 * k;
      float e = __expf(s_en[r * 128 + c] - rmax);
      s_en[r * 128 + c] = e;
      ps += e;
    }
    s_rsm[r * 16 + cg] = ps;
    __syncthreads();

    float rsum = 0.0f;
#pragma unroll
    for (int j = 0; j < 16; ++j) rsum += s_rsm[r * 16 + j];
    const float inv = fast_rcp(rsum);

    // ---- beta_new, streaming (non-temporal) outputs, carry bf16 beta ----
#pragma unroll
    for (int k = 0; k < 8; ++k) {
      const int c = cg + 16 * k;
      const float bnew = s_en[r * 128 + c] * inv;
      s_beta[r * 128 + c] = (__bf16)bnew;
      __builtin_nontemporal_store(
          bnew * xf[r * 128 + c],
          &out_sig[(size_t)t * (B_DIM * D_DIM) + (size_t)(m0 + r) * D_DIM + c]);
      __builtin_nontemporal_store(
          bnew,
          &out_beta[(size_t)(m0 + r) * (T_DIM * D_DIM) + (size_t)t * D_DIM + c]);
    }
    // no trailing barrier: next iteration's top barrier orders all LDS reuse
  }
}

extern "C" void kernel_launch(void* const* d_in, const int* in_sizes, int n_in,
                              void* d_out, int out_size, void* d_ws, size_t ws_size,
                              hipStream_t stream) {
  const float* x  = (const float*)d_in[0];
  const float* Wx = (const float*)d_in[1];
  const float* bx = (const float*)d_in[2];
  const float* Wb = (const float*)d_in[3];
  const float* bb = (const float*)d_in[4];
  const float* We = (const float*)d_in[5];
  const float* be = (const float*)d_in[6];
  float* out_sig  = (float*)d_out;
  float* out_beta = out_sig + (size_t)T_DIM * B_DIM * D_DIM;

  hipLaunchKernelGGL(sensor_attention_scan, dim3(B_DIM / 16), dim3(256), 0, stream,
                     x, Wx, bx, Wb, bb, We, be, out_sig, out_beta);
}
